// MultiHeadAttention_77429670413045
// MI455X (gfx1250) — matrix operand order repaired and checked by reading, not compiled, not run on hardware
//
#include <hip/hip_runtime.h>
#include <hip/hip_bf16.h>
#include <math.h>

// ---------------------------------------------------------------------------
// MI455X (gfx1250) multi-head attention
//   - all GEMMs via v_wmma_f32_16x16x32_f16 (f32 accumulate)
//   - K/V and weight tiles staged into LDS with global_load_async_to_lds_b128
//     (ASYNCcnt-tracked, double buffered, overlapped with WMMA compute)
// ---------------------------------------------------------------------------

typedef _Float16 half_t;
typedef __attribute__((ext_vector_type(16))) _Float16 v16h;
typedef __attribute__((ext_vector_type(2)))  _Float16 h2;
typedef __attribute__((ext_vector_type(8)))  float    v8f;

#define B_  4
#define S_  2048
#define D_  1024
#define H_  16
#define HD_ 64
#define PST 34   // LDS row stride for P-tile (pad to dodge bank conflicts)

// ---- CDNA5 async copy: global -> LDS, 16 bytes per lane, ASYNCcnt tracked --
static __device__ __forceinline__ unsigned lds_off(const void* p) {
  // AS3 offset == low 32 bits of the generic pointer (aperture form)
  return (unsigned)(unsigned long long)(uintptr_t)p;
}
static __device__ __forceinline__ void async_copy_b128(unsigned lds_byte_off,
                                                       const void* gaddr) {
  asm volatile("global_load_async_to_lds_b128 %0, %1, off"
               :
               : "v"(lds_byte_off), "v"((unsigned long long)(uintptr_t)gaddr)
               : "memory");
}
static __device__ __forceinline__ void wait_async0() {
  asm volatile("s_wait_asynccnt 0" ::: "memory");
}

// ---- WMMA fragment loaders (contiguous k = 16*half + s in A and B) ------

// A fragment, 16x32 f16, from fp32 row-major source (convert on the fly).
static __device__ __forceinline__ v16h load_a_f32(const float* __restrict__ src, int lda) {
  const int lane = threadIdx.x & 31;
  const int m = lane & 15, hf = lane >> 4;
  v16h a;
#pragma unroll
  for (int i = 0; i < 8; ++i) {
    const int kk = 16 * hf + 2 * i;
    const float2 t = *(const float2*)(src + (size_t)m * lda + kk);
    a[2 * i + 0] = (_Float16)t.x;
    a[2 * i + 1] = (_Float16)t.y;
  }
  return a;
}

// A fragment, 16x32 f16, from f16 row-major source (global or LDS).
static __device__ __forceinline__ v16h load_a_f16(const half_t* src, int lda) {
  const int lane = threadIdx.x & 31;
  const int m = lane & 15, hf = lane >> 4;
  v16h a;
#pragma unroll
  for (int i = 0; i < 8; ++i) {
    const int kk = 16 * hf + 2 * i;
    const h2 t = *(const h2*)(src + (size_t)m * lda + kk);
    a[2 * i + 0] = t.x;
    a[2 * i + 1] = t.y;
  }
  return a;
}

// B fragment, 32x16 f16, from N-major storage: element (k,n) at src[n*ldb + k].
static __device__ __forceinline__ v16h load_b_nmajor(const half_t* src, int ldb) {
  const int lane = threadIdx.x & 31;
  const int n = lane & 15, hf = lane >> 4;
  v16h b;
#pragma unroll
  for (int i = 0; i < 8; ++i) {
    const int kk = 16 * hf + 2 * i;
    const h2 t = *(const h2*)(src + (size_t)n * ldb + kk);
    b[2 * i + 0] = t.x;
    b[2 * i + 1] = t.y;
  }
  return b;
}

// B fragment, 32x16 f16, from K-major storage: element (k,n) at src[k*ldb + n].
static __device__ __forceinline__ v16h load_b_kmajor(const half_t* src, int ldb) {
  const int lane = threadIdx.x & 31;
  const int n = lane & 15, hf = lane >> 4;
  v16h b;
#pragma unroll
  for (int i = 0; i < 8; ++i) {
    const int kk = 16 * hf + 2 * i;
    b[2 * i + 0] = src[(size_t)(kk + 0) * ldb + n];
    b[2 * i + 1] = src[(size_t)(kk + 1) * ldb + n];
  }
  return b;
}

static __device__ __forceinline__ v8f wmma16(v16h a, v16h b, v8f c) {
  return __builtin_amdgcn_wmma_f32_16x16x32_f16(
      /*neg_a=*/false, a, /*neg_b=*/false, b,
      /*c_mod=*/(short)0, c, /*reuse_a=*/false, /*reuse_b=*/false);
}

// ---------------------------------------------------------------------------
// Weight pack: W[k][n] fp32 -> Wt[n][k] f16 (N-major so B-frags load h2 pairs)
// ---------------------------------------------------------------------------
__global__ void pack_wt(const float* __restrict__ W, half_t* __restrict__ Wt) {
  const int n = blockIdx.x * 16 + threadIdx.x;
  const int k = blockIdx.y * 16 + threadIdx.y;
  Wt[(size_t)n * D_ + k] = (_Float16)W[(size_t)k * D_ + n];
}

// ---------------------------------------------------------------------------
// Tiled GEMM: out[M,N] = A[M,K] @ W + bias.
// Block = 256 threads (8 waves) arranged 4x2 -> 128x128 block tile;
// wave tile 32x64 (2x4 C frags). Weight tile (128x32 f16, 8KB) is staged
// into LDS with async b128 copies, double buffered across the K loop.
// OUT_QKV: write f16 into [B,H,S,HD]; else write fp32 flat [M,N].
// ---------------------------------------------------------------------------
template <bool A_F16, bool OUT_QKV>
__global__ __launch_bounds__(256) void gemm_wmma(const void* __restrict__ A_,
                                                 const half_t* __restrict__ Wt,
                                                 const float* __restrict__ bias,
                                                 void* __restrict__ out,
                                                 int M, int N, int K) {
  __shared__ half_t bstage[2][128 * 32];      // [buf][n-major 128 x 32]
  const int lane = threadIdx.x & 31;
  const int wid  = threadIdx.x >> 5;
  const int wrow = wid >> 1;                  // 0..3
  const int wcol = wid & 1;                   // 0..1
  const int col0b = blockIdx.x * 128;
  const int row0  = blockIdx.y * 128 + wrow * 32;
  const int col0w = wcol * 64;                // within block tile

  // cooperative async stage of Wt[col0b..+127][k0..+31] -> bstage[buf]
  auto stage_w = [&](int k0, int buf) {
#pragma unroll
    for (int j = 0; j < 2; ++j) {
      const int c  = threadIdx.x * 2 + j;     // 512 x 16B chunks = 8KB
      const int n  = c >> 2;
      const int cs = (c & 3) * 8;             // 8 halfs = 16B
      async_copy_b128(lds_off(&bstage[buf][n * 32 + cs]),
                      Wt + (size_t)(col0b + n) * K + k0 + cs);
    }
  };

  v8f acc[2][4] = {};

  stage_w(0, 0);
  const int nK = K / 32;
  for (int it = 0; it < nK; ++it) {
    const int k0 = it * 32;
    wait_async0();                // this wave's copies for tile `it` landed
    __syncthreads();              // everyone's copies landed; prev readers done
    if (it + 1 < nK) stage_w(k0 + 32, (it + 1) & 1);

    v16h a[2];
#pragma unroll
    for (int mt = 0; mt < 2; ++mt) {
      if (A_F16)
        a[mt] = load_a_f16((const half_t*)A_ + (size_t)(row0 + mt * 16) * K + k0, K);
      else
        a[mt] = load_a_f32((const float*)A_ + (size_t)(row0 + mt * 16) * K + k0, K);
    }
    if (it + 1 < nK) {            // gfx1250 global_prefetch_b8 for next A tile
      if (A_F16) __builtin_prefetch((const half_t*)A_ + (size_t)row0 * K + k0 + 32, 0, 0);
      else       __builtin_prefetch((const float*)A_ + (size_t)row0 * K + k0 + 32, 0, 0);
    }

    const half_t* bt = &bstage[it & 1][0];
#pragma unroll
    for (int nt = 0; nt < 4; ++nt) {
      const v16h b = load_b_nmajor(bt + (col0w + nt * 16) * 32, 32);
#pragma unroll
      for (int mt = 0; mt < 2; ++mt) acc[mt][nt] = wmma16(a[mt], b, acc[mt][nt]);
    }
  }

  const int hf = lane >> 4, nl = lane & 15;
#pragma unroll
  for (int mt = 0; mt < 2; ++mt) {
#pragma unroll
    for (int nt = 0; nt < 4; ++nt) {
      const int col  = col0b + col0w + nt * 16 + nl;
      const float bb = bias[col];
#pragma unroll
      for (int r = 0; r < 8; ++r) {
        const int row = row0 + mt * 16 + r + 8 * hf;   // C layout: M = r + 8*half
        const float v = acc[mt][nt][r] + bb;
        if (OUT_QKV) {
          const int bi = row >> 11;          // row / S_
          const int s  = row & (S_ - 1);
          const int h  = col >> 6;           // col / HD_
          const int hd = col & (HD_ - 1);
          ((half_t*)out)[(((size_t)bi * H_ + h) * S_ + s) * HD_ + hd] = (_Float16)v;
        } else {
          ((float*)out)[(size_t)row * N + col] = v;
        }
      }
    }
  }
}

// ---------------------------------------------------------------------------
// Flash attention. Block = 8 waves, all on the SAME (b,h); wave w owns q-tile
// [q0, q0+32). K/V kv-tiles (32x64 f16 = 4KB each, contiguous in [B,H,S,HD])
// are staged once per block via async b128 copies, double buffered, so the
// 8 waves share one copy of K/V instead of 8 redundant global streams.
// ---------------------------------------------------------------------------
__global__ __launch_bounds__(256) void attn_wmma(const half_t* __restrict__ Q,
                                                 const half_t* __restrict__ Kt,
                                                 const half_t* __restrict__ V,
                                                 half_t* __restrict__ ctx) {
  __shared__ half_t kvb[2][2][32 * 64];   // [buf][0=K,1=V][kv 32 x hd 64]
  __shared__ half_t plds[8][32][PST];     // per-wave P transpose staging

  const int lane = threadIdx.x & 31;
  const int wid  = threadIdx.x >> 5;
  const int hf = lane >> 4, nl = lane & 15;
  const int bh = blockIdx.y;                       // b*H + h
  const int q0 = (blockIdx.x * 8 + wid) * 32;
  const size_t base = (size_t)bh * S_ * HD_;

  // cooperative async stage of one kv tile (K and V) -> kvb[buf]
  auto stage_kv = [&](int kv0, int buf) {
    const unsigned t = threadIdx.x * 8;           // 8 halfs = 16B per thread
    async_copy_b128(lds_off(&kvb[buf][0][t]), Kt + base + (size_t)kv0 * HD_ + t);
    async_copy_b128(lds_off(&kvb[buf][1][t]), V  + base + (size_t)kv0 * HD_ + t);
  };

  // Q A-fragments: 2 M-tiles x 2 K-chunks (HD=64)
  v16h aq[2][2];
#pragma unroll
  for (int mt = 0; mt < 2; ++mt)
#pragma unroll
    for (int kc = 0; kc < 2; ++kc)
      aq[mt][kc] = load_a_f16(Q + base + (size_t)(q0 + mt * 16) * HD_ + kc * 32, HD_);

  v8f cacc[2][4] = {};
  float mrun[2][8], lrun[2][8];
#pragma unroll
  for (int mt = 0; mt < 2; ++mt)
#pragma unroll
    for (int r = 0; r < 8; ++r) { mrun[mt][r] = -INFINITY; lrun[mt][r] = 0.0f; }

  const float scale = 0.125f;   // 1/sqrt(HD)

  stage_kv(0, 0);
  for (int it = 0; it < S_ / 32; ++it) {
    wait_async0();
    __syncthreads();
    if (it + 1 < S_ / 32) stage_kv((it + 1) * 32, (it + 1) & 1);
    const half_t* ktile = &kvb[it & 1][0][0];
    const half_t* vtile = &kvb[it & 1][1][0];

    // ---- scores tile S[32 x 32] = Q * K^T (K tile read N-major from LDS) --
    v8f sc[2][2] = {};
#pragma unroll
    for (int nt = 0; nt < 2; ++nt) {
#pragma unroll
      for (int kc = 0; kc < 2; ++kc) {
        const v16h bk = load_b_nmajor(ktile + (nt * 16) * HD_ + kc * 32, HD_);
#pragma unroll
        for (int mt = 0; mt < 2; ++mt) sc[mt][nt] = wmma16(aq[mt][kc], bk, sc[mt][nt]);
      }
    }

    // ---- online softmax over the 32 new columns ---------------------------
#pragma unroll
    for (int mt = 0; mt < 2; ++mt) {
#pragma unroll
      for (int r = 0; r < 8; ++r) {
        const float s0 = sc[mt][0][r] * scale;
        const float s1 = sc[mt][1][r] * scale;
        float vmax = fmaxf(s0, s1);
#pragma unroll
        for (int off = 1; off <= 8; off <<= 1)
          vmax = fmaxf(vmax, __shfl_xor(vmax, off, 32));   // stays in 16-lane half
        const float mnew = fmaxf(mrun[mt][r], vmax);
        const float corr = __expf(mrun[mt][r] - mnew);     // exp(-inf)=0 first pass
        mrun[mt][r] = mnew;
        const float p0 = __expf(s0 - mnew);
        const float p1 = __expf(s1 - mnew);
        sc[mt][0][r] = p0;
        sc[mt][1][r] = p1;
        float psum = p0 + p1;
#pragma unroll
        for (int off = 1; off <= 8; off <<= 1) psum += __shfl_xor(psum, off, 32);
        lrun[mt][r] = lrun[mt][r] * corr + psum;
#pragma unroll
        for (int nt = 0; nt < 4; ++nt) cacc[mt][nt][r] *= corr;
      }
    }

    // ---- transpose P: C layout -> A layout via per-wave LDS tile ----------
#pragma unroll
    for (int mt = 0; mt < 2; ++mt)
#pragma unroll
      for (int nt = 0; nt < 2; ++nt)
#pragma unroll
        for (int r = 0; r < 8; ++r)
          plds[wid][mt * 16 + r + 8 * hf][nt * 16 + nl] = (_Float16)sc[mt][nt][r];
    asm volatile("s_wait_dscnt 0" ::: "memory");   // DS in-order per wave; be explicit

    v16h ap[2];
#pragma unroll
    for (int mt = 0; mt < 2; ++mt) ap[mt] = load_a_f16(&plds[wid][mt * 16][0], PST);

    // ---- ctx += P * V (V tile read K-major from LDS) -----------------------
#pragma unroll
    for (int nt = 0; nt < 4; ++nt) {
      const v16h bv = load_b_kmajor(vtile + nt * 16, HD_);
#pragma unroll
      for (int mt = 0; mt < 2; ++mt) cacc[mt][nt] = wmma16(ap[mt], bv, cacc[mt][nt]);
    }
  }

  // ---- normalize and store ctx as f16 [B,S,D] -----------------------------
  const int bi = bh / H_;
  const int h  = bh % H_;
#pragma unroll
  for (int mt = 0; mt < 2; ++mt) {
#pragma unroll
    for (int r = 0; r < 8; ++r) {
      const float inv = 1.0f / lrun[mt][r];
      const int s = q0 + mt * 16 + r + 8 * hf;
#pragma unroll
      for (int nt = 0; nt < 4; ++nt) {
        const int d = h * HD_ + nt * 16 + nl;
        ctx[((size_t)bi * S_ + s) * D_ + d] = (_Float16)(cacc[mt][nt][r] * inv);
      }
    }
  }
}

// ---------------------------------------------------------------------------
// Launch: pack weights -> Q/K/V projections -> flash attention -> out proj
// ---------------------------------------------------------------------------
extern "C" void kernel_launch(void* const* d_in, const int* in_sizes, int n_in,
                              void* d_out, int out_size, void* d_ws, size_t ws_size,
                              hipStream_t stream) {
  (void)in_sizes; (void)n_in; (void)out_size; (void)ws_size;
  const float* query = (const float*)d_in[0];
  const float* key_  = (const float*)d_in[1];
  const float* value = (const float*)d_in[2];
  const float* wq    = (const float*)d_in[3];
  const float* bq    = (const float*)d_in[4];
  const float* wk    = (const float*)d_in[5];
  const float* bk    = (const float*)d_in[6];
  const float* wv    = (const float*)d_in[7];
  const float* bv    = (const float*)d_in[8];
  const float* wo    = (const float*)d_in[9];
  const float* bo    = (const float*)d_in[10];

  char* ws = (char*)d_ws;
  const size_t WB = (size_t)D_ * D_ * sizeof(half_t);        // 2 MB / weight
  const size_t TB = (size_t)B_ * S_ * D_ * sizeof(half_t);   // 16 MB / tensor
  half_t* wqt = (half_t*)(ws + 0 * WB);
  half_t* wkt = (half_t*)(ws + 1 * WB);
  half_t* wvt = (half_t*)(ws + 2 * WB);
  half_t* wot = (half_t*)(ws + 3 * WB);
  half_t* Qh  = (half_t*)(ws + 4 * WB + 0 * TB);
  half_t* Kh  = (half_t*)(ws + 4 * WB + 1 * TB);
  half_t* Vh  = (half_t*)(ws + 4 * WB + 2 * TB);
  half_t* Ch  = (half_t*)(ws + 4 * WB + 3 * TB);

  dim3 pg(D_ / 16, D_ / 16), pb(16, 16);
  pack_wt<<<pg, pb, 0, stream>>>(wq, wqt);
  pack_wt<<<pg, pb, 0, stream>>>(wk, wkt);
  pack_wt<<<pg, pb, 0, stream>>>(wv, wvt);
  pack_wt<<<pg, pb, 0, stream>>>(wo, wot);

  const int M = B_ * S_, N = D_, K = D_;
  dim3 gg(N / 128, M / 128), gb(256);
  gemm_wmma<false, true><<<gg, gb, 0, stream>>>(query, wqt, bq, Qh, M, N, K);
  gemm_wmma<false, true><<<gg, gb, 0, stream>>>(key_,  wkt, bk, Kh, M, N, K);
  gemm_wmma<false, true><<<gg, gb, 0, stream>>>(value, wvt, bv, Vh, M, N, K);

  dim3 ag(S_ / (32 * 8), B_ * H_), ab(256);
  attn_wmma<<<ag, ab, 0, stream>>>(Qh, Kh, Vh, Ch);

  gemm_wmma<true, false><<<gg, gb, 0, stream>>>(Ch, wot, bo, (float*)d_out, M, N, K);
}
